// LittleLanguageModel_12335146074302
// MI455X (gfx1250) — matrix operand order repaired and checked by reading, not compiled, not run on hardware
//
#include <hip/hip_runtime.h>
#include <hip/hip_bf16.h>
#include <math.h>

typedef __bf16 bf16;
typedef __attribute__((ext_vector_type(16))) __bf16 v16bf;
typedef __attribute__((ext_vector_type(2)))  __bf16 bf16x2;
typedef __attribute__((ext_vector_type(8)))  float   v8f;
typedef __attribute__((ext_vector_type(4)))  float   f4;

union AFrag { v16bf v; f4 q[2]; };

__device__ inline v8f vzero8() {
  v8f z;
#pragma unroll
  for (int i = 0; i < 8; ++i) z[i] = 0.0f;
  return z;
}

// Async copy: global (bf16, 16B) -> LDS, tracked by ASYNCcnt (CDNA5 path).
// VDST = LDS byte address (low 32 bits of generic pointer), VADDR = 64-bit global.
__device__ inline void async_copy_b128(const void* lds_dst, const void* gsrc) {
  unsigned lds_addr = (unsigned)(size_t)lds_dst;
  asm volatile("global_load_async_to_lds_b128 %0, %1, off"
               :: "v"(lds_addr), "v"(gsrc)
               : "memory");
}
__device__ inline void wait_async0() {
  asm volatile("s_wait_asynccnt 0" ::: "memory");
}

// ---------------------------------------------------------------------------
// Embedding: x[b*T+t, :] = tok_emb[idx[b*T+t], :] + pos_emb[t, :]
// ---------------------------------------------------------------------------
__global__ __launch_bounds__(256)
void embed_kernel(const int* __restrict__ idx, const float* __restrict__ tok,
                  const float* __restrict__ pos, float* __restrict__ x)
{
  int row = blockIdx.x;            // 0..4095
  int t   = row & 2047;
  int tid = threadIdx.x;           // 256 threads * 4 floats = 1024
  int token = idx[row];
  f4 a = reinterpret_cast<const f4*>(tok + (size_t)token * 1024)[tid];
  f4 p = reinterpret_cast<const f4*>(pos + (size_t)t * 1024)[tid];
  reinterpret_cast<f4*>(x + (size_t)row * 1024)[tid] = a + p;
}

// ---------------------------------------------------------------------------
// LayerNorm over E=1024, f32 in -> bf16 out (feeds WMMA A operands)
// ---------------------------------------------------------------------------
__global__ __launch_bounds__(256)
void layernorm_bf16(const float* __restrict__ x, const float* __restrict__ g,
                    const float* __restrict__ bta, bf16* __restrict__ out)
{
  int row = blockIdx.x, tid = threadIdx.x;
  int lane = tid & 31, wid = tid >> 5;
  const float* xr = x + (size_t)row * 1024;
  f4 xv = reinterpret_cast<const f4*>(xr)[tid];
  float s  = xv[0] + xv[1] + xv[2] + xv[3];
  float s2 = xv[0]*xv[0] + xv[1]*xv[1] + xv[2]*xv[2] + xv[3]*xv[3];
#pragma unroll
  for (int d = 1; d < 32; d <<= 1) {
    s  += __shfl_xor(s,  d, 32);
    s2 += __shfl_xor(s2, d, 32);
  }
  __shared__ float rs[8], rs2[8], mb[2];
  if (lane == 0) { rs[wid] = s; rs2[wid] = s2; }
  __syncthreads();
  if (tid == 0) {
    float a = 0.f, b2 = 0.f;
    for (int i = 0; i < 8; ++i) { a += rs[i]; b2 += rs2[i]; }
    float mean = a * (1.0f / 1024.0f);
    float var  = b2 * (1.0f / 1024.0f) - mean * mean;
    mb[0] = mean; mb[1] = rsqrtf(var + 1e-5f);
  }
  __syncthreads();
  float mean = mb[0], inv = mb[1];
  f4 gv = reinterpret_cast<const f4*>(g)[tid];
  f4 bv = reinterpret_cast<const f4*>(bta)[tid];
  int c = tid * 4;
  bf16* orow = out + (size_t)row * 1024;
#pragma unroll
  for (int u = 0; u < 4; ++u)
    orow[c + u] = (bf16)((xv[u] - mean) * inv * gv[u] + bv[u]);
}

// ---------------------------------------------------------------------------
// Generic WMMA GEMM:  out = op(A_bf16[M,K] @ W_f32[K,N] + bias + res)
// 128x128 block tile, 8 waves of 64x32, K-step 32, double-buffered LDS.
// A tile: global_load_async_to_lds_b128 (ASYNCcnt). W tile: f32 loads,
// packed cvt to bf16x2, transposed [n][k] LDS layout.
// ---------------------------------------------------------------------------
#define BM 128
#define BN 128
#define BK 32

__global__ __launch_bounds__(256)
void gemm_bf16(const bf16* __restrict__ A, const float* __restrict__ W,
               const float* __restrict__ bias, const float* __restrict__ res,
               float* __restrict__ outF, bf16* __restrict__ outB,
               int M, int N, int K, int relu)
{
  (void)M;
  __shared__ __align__(16) bf16 As[2][BM][BK + 8];   // row-major [m][k]
  __shared__ __align__(16) bf16 Bs[2][BN][BK + 8];   // transposed [n][k]

  int tid  = threadIdx.x;
  int lane = tid & 31, wid = tid >> 5;
  int lm = lane & 15, lh = lane >> 4;
  int m0 = blockIdx.y * BM, n0 = blockIdx.x * BN;
  int wm = (wid >> 2) * 64;      // wave M offset within block tile
  int wn = (wid & 3) * 32;       // wave N offset

  // staging assignments
  int arow = tid >> 1, ahalf = (tid & 1) * 16;       // A: 32B per thread
  int bkp  = tid >> 4;                               // W: k-row pair 0..15
  int bns  = (tid & 15) * 8;                         // W: 8 columns

  v8f acc[4][2];
#pragma unroll
  for (int i = 0; i < 4; ++i)
#pragma unroll
    for (int j = 0; j < 2; ++j) acc[i][j] = vzero8();

  auto stage = [&](int buf, int k0) {
    // --- A tile: async straight to LDS ---
    const bf16* ga = A + (size_t)(m0 + arow) * K + k0 + ahalf;
    async_copy_b128(&As[buf][arow][ahalf], ga);
    async_copy_b128(&As[buf][arow][ahalf + 8], ga + 8);
    // --- W tile: two k-rows, 8 columns; pack (k,k+1) into bf16x2 dwords ---
    const f4* r0 = reinterpret_cast<const f4*>(W + (size_t)(k0 + 2 * bkp)     * N + n0 + bns);
    const f4* r1 = reinterpret_cast<const f4*>(W + (size_t)(k0 + 2 * bkp + 1) * N + n0 + bns);
    f4 w00 = r0[0], w01 = r0[1];
    f4 w10 = r1[0], w11 = r1[1];
#pragma unroll
    for (int u = 0; u < 4; ++u) {
      bf16x2 pk; pk[0] = (bf16)w00[u]; pk[1] = (bf16)w10[u];
      *reinterpret_cast<bf16x2*>(&Bs[buf][bns + u][2 * bkp]) = pk;
    }
#pragma unroll
    for (int u = 0; u < 4; ++u) {
      bf16x2 pk; pk[0] = (bf16)w01[u]; pk[1] = (bf16)w11[u];
      *reinterpret_cast<bf16x2*>(&Bs[buf][bns + 4 + u][2 * bkp]) = pk;
    }
  };

  stage(0, 0);
  int nk = K / BK;
  for (int i = 0; i < nk; ++i) {
    int buf = i & 1;
    wait_async0();         // this wave's async A writes (buffer i) done
    __syncthreads();       // all waves' tile writes visible
    if (i + 1 < nk) {
      __builtin_prefetch(W + (size_t)((i + 1) * BK + 2 * bkp) * N + n0 + bns, 0, 0);
      stage(buf ^ 1, (i + 1) * BK);   // overlap next tile with this compute
    }

    // --- fragments (ISA 7.12.2 layouts) ---
    AFrag a[4], b[2];
#pragma unroll
    for (int mt = 0; mt < 4; ++mt) {
      const bf16* p = &As[buf][wm + mt * 16 + lm][0];
      a[mt].q[0] = *reinterpret_cast<const f4*>(p + lh * 8);        // K 0..7 / 8..15
      a[mt].q[1] = *reinterpret_cast<const f4*>(p + 16 + lh * 8);   // K 16..23 / 24..31
    }
#pragma unroll
    for (int nt = 0; nt < 2; ++nt) {
      const bf16* p = &Bs[buf][wn + nt * 16 + lm][0];
      b[nt].q[0] = *reinterpret_cast<const f4*>(p + lh * 8);        // K 0..7 / 8..15
      b[nt].q[1] = *reinterpret_cast<const f4*>(p + 16 + lh * 8);   // K 16..23 / 24..31
    }
#pragma unroll
    for (int nt = 0; nt < 2; ++nt)
#pragma unroll
      for (int mt = 0; mt < 4; ++mt)
        acc[mt][nt] = __builtin_amdgcn_wmma_f32_16x16x32_bf16(
            false, a[mt].v, false, b[nt].v, (short)0, acc[mt][nt], false, false);
  }

  // --- epilogue: bias / residual / relu / f32-or-bf16 store ---
  float bia[2] = {0.0f, 0.0f};
  if (bias) {
    bia[0] = bias[n0 + wn + lm];
    bia[1] = bias[n0 + wn + 16 + lm];
  }
#pragma unroll
  for (int mt = 0; mt < 4; ++mt)
#pragma unroll
    for (int nt = 0; nt < 2; ++nt)
#pragma unroll
      for (int r = 0; r < 8; ++r) {
        int m = m0 + wm + mt * 16 + lh * 8 + r;
        int n = n0 + wn + nt * 16 + lm;
        float val = acc[mt][nt][r] + bia[nt];
        if (res)  val += res[(size_t)m * N + n];
        if (relu) val = fmaxf(val, 0.0f);
        if (outB) outB[(size_t)m * N + n] = (bf16)val;
        else      outF[(size_t)m * N + n] = val;
      }
}

// ---------------------------------------------------------------------------
// Flash attention, causal, H=16 heads of D=64, T=2048.
// Block = 4 waves = 64 q rows; each wave owns a 16-row Q tile.
// Per 32-key step: 4 WMMAs for S = Q*K^T, online softmax, 4 WMMAs for P*V.
// K tile staged via async-to-LDS; V tile transposed through VGPRs.
// ---------------------------------------------------------------------------
__global__ __launch_bounds__(128)
void attn_kernel(const bf16* __restrict__ q, const bf16* __restrict__ k,
                 const bf16* __restrict__ v, bf16* __restrict__ o)
{
  const int T = 2048, E = 1024, Dh = 64;
  int qblk = blockIdx.x, h = blockIdx.y, b = blockIdx.z;
  int tid = threadIdx.x, lane = tid & 31, wave = tid >> 5;
  int lm = lane & 15, lh = lane >> 4;
  int qbase = qblk * 64 + wave * 16;
  size_t rowbase = (size_t)b * T;
  int col = h * Dh;

  __shared__ __align__(16) bf16 Ks[32][72];       // [key][d]
  __shared__ __align__(16) bf16 Vs[64][40];       // [d][key]  (transposed)
  __shared__ __align__(16) bf16 Ps[4][16][40];    // per-wave P re-layout scratch

  // Q fragments (A layout, 16x32 per D-chunk), loaded once
  AFrag qa[2];
  {
    const bf16* qrow = q + (rowbase + qbase + lm) * E + col;
#pragma unroll
    for (int c = 0; c < 2; ++c) {
      qa[c].q[0] = *reinterpret_cast<const f4*>(qrow + c * 32 + lh * 8);
      qa[c].q[1] = *reinterpret_cast<const f4*>(qrow + c * 32 + 16 + lh * 8);
    }
  }

  v8f oacc[4];
#pragma unroll
  for (int nt = 0; nt < 4; ++nt) oacc[nt] = vzero8();
  float mst[8], lst[8];
#pragma unroll
  for (int r = 0; r < 8; ++r) { mst[r] = -1e30f; lst[r] = 0.0f; }

  int jsteps = qblk * 2 + 2;                 // uniform across block (causal)
  for (int js = 0; js < jsteps; ++js) {
    int kbase = js * 32;
    // cooperative stage: K async to LDS, V transposed through VGPRs
    {
      int r = tid >> 2, seg = (tid & 3) * 16;
      const bf16* ksrc = k + (rowbase + kbase + r) * E + col + seg;
      async_copy_b128(&Ks[r][seg], ksrc);
      async_copy_b128(&Ks[r][seg + 8], ksrc + 8);
      const f4* vsrc = reinterpret_cast<const f4*>(v + (rowbase + kbase + r) * E + col + seg);
      union { f4 qq[2]; bf16 e[16]; } vt;
      vt.qq[0] = vsrc[0]; vt.qq[1] = vsrc[1];
#pragma unroll
      for (int i = 0; i < 16; ++i) Vs[seg + i][r] = vt.e[i];
    }
    wait_async0();
    __syncthreads();

    // S = Q * K^T  (two 16-key subtiles, two D-chunks each)
    v8f s0 = vzero8(), s1 = vzero8();
#pragma unroll
    for (int c = 0; c < 2; ++c) {
      AFrag kb0, kb1;
      const bf16* kr0 = &Ks[lm][0] + c * 32;
      kb0.q[0] = *reinterpret_cast<const f4*>(kr0 + lh * 8);
      kb0.q[1] = *reinterpret_cast<const f4*>(kr0 + 16 + lh * 8);
      const bf16* kr1 = &Ks[16 + lm][0] + c * 32;
      kb1.q[0] = *reinterpret_cast<const f4*>(kr1 + lh * 8);
      kb1.q[1] = *reinterpret_cast<const f4*>(kr1 + 16 + lh * 8);
      s0 = __builtin_amdgcn_wmma_f32_16x16x32_bf16(false, qa[c].v, false, kb0.v, (short)0, s0, false, false);
      s1 = __builtin_amdgcn_wmma_f32_16x16x32_bf16(false, qa[c].v, false, kb1.v, (short)0, s1, false, false);
    }

    // online softmax (row m = lh*8 + r lives across the 16 lanes of a half)
#pragma unroll
    for (int r = 0; r < 8; ++r) {
      int m = lh * 8 + r;
      int qi = qbase + m;
      float a0 = s0[r] * 0.125f;
      float a1 = s1[r] * 0.125f;
      if (kbase + lm > qi)      a0 = -1e30f;
      if (kbase + 16 + lm > qi) a1 = -1e30f;
      float t = fmaxf(a0, a1);
#pragma unroll
      for (int d = 1; d < 16; d <<= 1) t = fmaxf(t, __shfl_xor(t, d, 32));
      float mn = fmaxf(mst[r], t);
      float f  = __expf(mst[r] - mn);
      float p0 = __expf(a0 - mn);
      float p1 = __expf(a1 - mn);
      float rsum = p0 + p1;
#pragma unroll
      for (int d = 1; d < 16; d <<= 1) rsum += __shfl_xor(rsum, d, 32);
      lst[r] = lst[r] * f + rsum;
      mst[r] = mn;
#pragma unroll
      for (int nt = 0; nt < 4; ++nt) oacc[nt][r] *= f;
      Ps[wave][m][lm]      = (bf16)p0;
      Ps[wave][m][16 + lm] = (bf16)p1;
    }

    // O += P @ V   (P re-read in A-fragment layout from wave-private LDS)
    AFrag pa;
    {
      const bf16* pr = &Ps[wave][lm][0];
      pa.q[0] = *reinterpret_cast<const f4*>(pr + lh * 8);
      pa.q[1] = *reinterpret_cast<const f4*>(pr + 16 + lh * 8);
    }
#pragma unroll
    for (int nt = 0; nt < 4; ++nt) {
      AFrag vb;
      const bf16* vr = &Vs[nt * 16 + lm][0];
      vb.q[0] = *reinterpret_cast<const f4*>(vr + lh * 8);
      vb.q[1] = *reinterpret_cast<const f4*>(vr + 16 + lh * 8);
      oacc[nt] = __builtin_amdgcn_wmma_f32_16x16x32_bf16(false, pa.v, false, vb.v, (short)0, oacc[nt], false, false);
    }
    __syncthreads();
  }

  // normalize and emit bf16 attention output
#pragma unroll
  for (int nt = 0; nt < 4; ++nt)
#pragma unroll
    for (int r = 0; r < 8; ++r) {
      int m = lh * 8 + r;
      float val = oacc[nt][r] / lst[r];
      o[(rowbase + qbase + m) * E + col + nt * 16 + lm] = (bf16)val;
    }
}

// ---------------------------------------------------------------------------
// Host orchestration
// ---------------------------------------------------------------------------
extern "C" void kernel_launch(void* const* d_in, const int* in_sizes, int n_in,
                              void* d_out, int out_size, void* d_ws, size_t ws_size,
                              hipStream_t stream)
{
  (void)in_sizes; (void)n_in; (void)out_size; (void)ws_size;
  const int Bsz = 2, T = 2048, E = 1024, Hh = 16, Vv = 32000, Lcnt = 8;
  const int M = Bsz * T;                    // 4096 rows

  const int*   idx  = (const int*)  d_in[0];
  const float* tok  = (const float*)d_in[1];
  const float* pos  = (const float*)d_in[2];
  const float* Wq   = (const float*)d_in[3];
  const float* Wk   = (const float*)d_in[4];
  const float* Wv   = (const float*)d_in[5];
  const float* Wo   = (const float*)d_in[6];
  const float* bo   = (const float*)d_in[7];
  const float* ln1g = (const float*)d_in[8];
  const float* ln1b = (const float*)d_in[9];
  const float* ln2g = (const float*)d_in[10];
  const float* ln2b = (const float*)d_in[11];
  const float* W1   = (const float*)d_in[12];
  const float* b1   = (const float*)d_in[13];
  const float* W2   = (const float*)d_in[14];
  const float* b2   = (const float*)d_in[15];
  const float* lnfg = (const float*)d_in[16];
  const float* lnfb = (const float*)d_in[17];
  const float* Wout = (const float*)d_in[18];
  const float* bout = (const float*)d_in[19];

  char* ws = (char*)d_ws;
  size_t off = 0;
  auto alloc = [&](size_t bytes) -> char* {
    char* p = ws + off;
    off += (bytes + 255) & ~(size_t)255;
    return p;
  };
  float* xA  = (float*)alloc((size_t)M * E * 4);
  float* xB  = (float*)alloc((size_t)M * E * 4);
  bf16*  hb  = (bf16*) alloc((size_t)M * E * 2);
  bf16*  qb  = (bf16*) alloc((size_t)M * E * 2);
  bf16*  kb  = (bf16*) alloc((size_t)M * E * 2);
  bf16*  vb  = (bf16*) alloc((size_t)M * E * 2);
  bf16*  ob  = (bf16*) alloc((size_t)M * E * 2);
  bf16*  mlp = (bf16*) alloc((size_t)M * 4 * E * 2);

  dim3 blk256(256), blk128(128);
  dim3 gE(E / BN, M / BM);          // (8, 32)   N=1024 GEMMs
  dim3 g4(4 * E / BN, M / BM);      // (32, 32)  N=4096 GEMM
  dim3 gO(Vv / BN, M / BM);         // (250, 32) logits GEMM
  dim3 gA(T / 64, Hh, Bsz);         // attention

  embed_kernel<<<M, blk256, 0, stream>>>(idx, tok, pos, xA);

  for (int l = 0; l < Lcnt; ++l) {
    const float* wq = Wq + (size_t)l * E * E;
    const float* wk = Wk + (size_t)l * E * E;
    const float* wv = Wv + (size_t)l * E * E;
    const float* wo = Wo + (size_t)l * E * E;
    const float* w1 = W1 + (size_t)l * E * 4 * E;
    const float* w2 = W2 + (size_t)l * 4 * E * E;

    layernorm_bf16<<<M, blk256, 0, stream>>>(xA, ln1g + (size_t)l * E, ln1b + (size_t)l * E, hb);
    gemm_bf16<<<gE, blk256, 0, stream>>>(hb, wq, nullptr, nullptr, nullptr, qb, M, E, E, 0);
    gemm_bf16<<<gE, blk256, 0, stream>>>(hb, wk, nullptr, nullptr, nullptr, kb, M, E, E, 0);
    gemm_bf16<<<gE, blk256, 0, stream>>>(hb, wv, nullptr, nullptr, nullptr, vb, M, E, E, 0);
    attn_kernel<<<gA, blk128, 0, stream>>>(qb, kb, vb, ob);
    gemm_bf16<<<gE, blk256, 0, stream>>>(ob, wo, bo + (size_t)l * E, xA, xB, nullptr, M, E, E, 0);
    layernorm_bf16<<<M, blk256, 0, stream>>>(xB, ln2g + (size_t)l * E, ln2b + (size_t)l * E, hb);
    gemm_bf16<<<g4, blk256, 0, stream>>>(hb, w1, b1 + (size_t)l * 4 * E, nullptr, nullptr, mlp, M, 4 * E, E, 1);
    gemm_bf16<<<gE, blk256, 0, stream>>>(mlp, w2, b2 + (size_t)l * E, xB, xA, nullptr, M, E, 4 * E, 0);
  }

  layernorm_bf16<<<M, blk256, 0, stream>>>(xA, lnfg, lnfb, hb);
  gemm_bf16<<<gO, blk256, 0, stream>>>(hb, Wout, bout, nullptr, (float*)d_out, nullptr, M, Vv, E, 0);
}
